// MultiHeadAttention_7662221656421
// MI455X (gfx1250) — compile-verified, matrix operand order checked by reading, not run on hardware
//
#include <hip/hip_runtime.h>

#define EMB   1024
#define HEADS 16
#define HD    64
#define BATCH 4
#define SEQ   2048
#define MTOT  (BATCH * SEQ)   // 8192

#define BM 128
#define BN 128
#define BK 32

typedef __attribute__((ext_vector_type(16))) _Float16     v16h;
typedef __attribute__((ext_vector_type(8)))  _Float16     v8h;
typedef __attribute__((ext_vector_type(8)))  float        v8f;
typedef __attribute__((ext_vector_type(4)))  unsigned int v4u;
typedef __attribute__((ext_vector_type(8)))  int          v8i;
typedef __attribute__((ext_vector_type(4)))  int          v4i;

#if defined(__has_builtin)
#if __has_builtin(__builtin_amdgcn_tensor_load_to_lds)
#define HAVE_TDM 1
#endif
#endif

__device__ __forceinline__ v8f wmma_f16(v16h a, v16h b, v8f c) {
  // D = A(16x32 f16) * B(32x16 f16, fed as 16x32 B^T layout) + C(16x16 f32)
  return __builtin_amdgcn_wmma_f32_16x16x32_f16(false, a, false, b, (short)0, c,
                                                false, false);
}

// Load a 16x32 f16 fragment (A layout; B operand uses the same layout with
// "row" meaning the output column).  stride in elements (halfs).
// Per ISA 7.12.2: lanes 0-15 hold rows 0-15 with K = {0..7, 16..23},
// lanes 16-31 hold rows 0-15 with K = {8..15, 24..31}.
template <typename PT>
__device__ __forceinline__ v16h load_frag(PT base, int stride, int k0) {
  const int lane = threadIdx.x & 31;
  const int r    = lane & 15;
  const int kk   = k0 + ((lane >> 4) << 3);
  const _Float16* p = (const _Float16*)base + (size_t)r * stride + kk;
  v8h lo = *(const v8h*)(p);
  v8h hi = *(const v8h*)(p + 16);
  v16h f;
#pragma unroll
  for (int i = 0; i < 8; ++i) { f[i] = lo[i]; f[i + 8] = hi[i]; }
  return f;
}

#ifdef HAVE_TDM
// Issue a TDM 2D tile load (f16 elements) into LDS.  Descriptor bitfields per
// CDNA5 ISA 08_async_tensor §8.3/8.4: Group0 = {count/type, lds_addr,
// global_addr}, Group1 = {data_size, tensor dims, tile dims, strides}.
// 2D tensor -> groups 2/3 disabled (zero; tile_dim2 = 0).
__device__ __forceinline__ void tdm_load_tile_f16(const _Float16* gsrc,
                                                  unsigned lds_off,
                                                  unsigned tile_rows,
                                                  unsigned tile_cols,
                                                  unsigned row_len,
                                                  unsigned tensor_rows,
                                                  unsigned row_stride) {
  const unsigned long long ga = (unsigned long long)(uintptr_t)gsrc;
  v4u g0;
  g0[0] = 1u;                                            // count=1, user mode
  g0[1] = lds_off;                                       // lds_addr (bytes)
  g0[2] = (unsigned)(ga & 0xffffffffull);                // global_addr lo
  g0[3] = (unsigned)((ga >> 32) & 0x01ffffffull) | (2u << 30);  // type=2
  v8i g1;
  g1[0] = (int)(1u << 16);                               // data_size=1 (2B)
  g1[1] = (int)((row_len & 0xffffu) << 16);              // tensor_dim0[15:0]
  g1[2] = (int)(((row_len >> 16) & 0xffffu) |
                ((tensor_rows & 0xffffu) << 16));        // dim0 hi | dim1 lo
  g1[3] = (int)(((tensor_rows >> 16) & 0xffffu) |
                (tile_cols << 16));                      // dim1 hi | tile_dim0
  g1[4] = (int)tile_rows;                                // tile_dim1 (dim2=0)
  g1[5] = (int)row_stride;                               // dim0_stride[31:0]
  g1[6] = 0;                                             // stride hi / dim1_stride
  g1[7] = 0;
  v4i gz = (v4i){0, 0, 0, 0};
#if __clang_major__ >= 23
  v8i gz8 = (v8i){0, 0, 0, 0, 0, 0, 0, 0};
  __builtin_amdgcn_tensor_load_to_lds(g0, g1, gz, gz, gz8, 0);
#else
  __builtin_amdgcn_tensor_load_to_lds(g0, g1, gz, gz, 0);
#endif
}
#endif  // HAVE_TDM

// ---------------------------------------------------------------------------
// Kernel 1: fused QKV projection.  Y = X @ W^T + b, f16 outputs in attention
// layouts.  blockIdx.z selects q/k/v:
//   z=0 -> Q  [B,H,S,Hd]   z=1 -> K  [B,H,S,Hd]   z=2 -> V^T[B,H,Hd,S]
// ---------------------------------------------------------------------------
__global__ __launch_bounds__(256) void qkv_proj_kernel(
    const float* __restrict__ q, const float* __restrict__ k,
    const float* __restrict__ v,
    const float* __restrict__ Wq, const float* __restrict__ bq,
    const float* __restrict__ Wk, const float* __restrict__ bk,
    const float* __restrict__ Wv, const float* __restrict__ bv,
    _Float16* __restrict__ Qh, _Float16* __restrict__ Kh,
    _Float16* __restrict__ Vt)
{
  const int z = blockIdx.z;
  const float* X    = (z == 0) ? q  : (z == 1) ? k  : v;
  const float* W    = (z == 0) ? Wq : (z == 1) ? Wk : Wv;
  const float* bias = (z == 0) ? bq : (z == 1) ? bk : bv;
  _Float16*    out  = (z == 0) ? Qh : (z == 1) ? Kh : Vt;

  __shared__ alignas(16) _Float16 Xs[BM][BK];
  __shared__ alignas(16) _Float16 Ws[BN][BK];

  const int tid  = threadIdx.x;
  const int wave = tid >> 5;
  const int lane = tid & 31;
  const int m0 = blockIdx.x * BM;
  const int n0 = blockIdx.y * BN;
  const int wm = (wave >> 2) * 64;
  const int wn = (wave & 3) * 32;

  v8f acc[4][2];
#pragma unroll
  for (int mt = 0; mt < 4; ++mt)
#pragma unroll
    for (int nt = 0; nt < 2; ++nt) acc[mt][nt] = (v8f){};

  const int r = tid >> 1;
  const int c = (tid & 1) * 16;

  for (int kt = 0; kt < EMB; kt += BK) {
    {
      const float4* src = (const float4*)(X + (size_t)(m0 + r) * EMB + kt + c);
#pragma unroll
      for (int i = 0; i < 2; ++i) {
        float4 f0 = src[2 * i], f1 = src[2 * i + 1];
        v8h hp;
        hp[0] = (_Float16)f0.x; hp[1] = (_Float16)f0.y;
        hp[2] = (_Float16)f0.z; hp[3] = (_Float16)f0.w;
        hp[4] = (_Float16)f1.x; hp[5] = (_Float16)f1.y;
        hp[6] = (_Float16)f1.z; hp[7] = (_Float16)f1.w;
        *(v8h*)&Xs[r][c + i * 8] = hp;
      }
      const float4* srw = (const float4*)(W + (size_t)(n0 + r) * EMB + kt + c);
#pragma unroll
      for (int i = 0; i < 2; ++i) {
        float4 f0 = srw[2 * i], f1 = srw[2 * i + 1];
        v8h hp;
        hp[0] = (_Float16)f0.x; hp[1] = (_Float16)f0.y;
        hp[2] = (_Float16)f0.z; hp[3] = (_Float16)f0.w;
        hp[4] = (_Float16)f1.x; hp[5] = (_Float16)f1.y;
        hp[6] = (_Float16)f1.z; hp[7] = (_Float16)f1.w;
        *(v8h*)&Ws[r][c + i * 8] = hp;
      }
    }
    __syncthreads();

    v16h bfr[2];
#pragma unroll
    for (int nt = 0; nt < 2; ++nt)
      bfr[nt] = load_frag(&Ws[wn + nt * 16][0], BK, 0);
#pragma unroll
    for (int mt = 0; mt < 4; ++mt) {
      v16h a = load_frag(&Xs[wm + mt * 16][0], BK, 0);
#pragma unroll
      for (int nt = 0; nt < 2; ++nt)
        acc[mt][nt] = wmma_f16(a, bfr[nt], acc[mt][nt]);
    }
    __syncthreads();
  }

  const int hi8 = (lane >> 4) << 3;
  const int col = lane & 15;
#pragma unroll
  for (int mt = 0; mt < 4; ++mt) {
#pragma unroll
    for (int nt = 0; nt < 2; ++nt) {
      const int n  = n0 + wn + nt * 16 + col;
      const float bval = bias[n];
      const int h  = n >> 6;
      const int hd = n & 63;
#pragma unroll
      for (int i = 0; i < 8; ++i) {
        const int m  = m0 + wm + mt * 16 + hi8 + i;
        const int b_ = m >> 11;
        const int s  = m & (SEQ - 1);
        const float val = acc[mt][nt][i] + bval;
        size_t idx;
        if (z == 2) idx = (((size_t)(b_ * HEADS + h)) * HD + hd) * SEQ + s;
        else        idx = (((size_t)(b_ * HEADS + h)) * SEQ + s) * HD + hd;
        out[idx] = (_Float16)val;
      }
    }
  }
}

// ---------------------------------------------------------------------------
// Kernel 2: flash attention.  One (b,h, 128-query tile) per block; each of 8
// waves owns 16 queries independently (online softmax, base-2).
// ---------------------------------------------------------------------------
#define SCALE_LOG2E 0.18033688f  // (1/sqrt(64)) * log2(e)

__global__ __launch_bounds__(256) void attn_kernel(
    const _Float16* __restrict__ Qh, const _Float16* __restrict__ Kh,
    const _Float16* __restrict__ Vt, _Float16* __restrict__ Ah)
{
  __shared__ alignas(16) _Float16 Ps[8][16][64];   // per-wave P tile

  const int bh   = blockIdx.y;
  const int b_   = bh >> 4;
  const int h    = bh & 15;
  const int wave = threadIdx.x >> 5;
  const int lane = threadIdx.x & 31;
  const int qBase = blockIdx.x * 128 + wave * 16;

  const _Float16* Qp = Qh + ((size_t)bh * SEQ + qBase) * HD;
  const _Float16* Kp = Kh + (size_t)bh * SEQ * HD;
  const _Float16* Vp = Vt + (size_t)bh * HD * SEQ;

  v16h qf[2];
  qf[0] = load_frag(Qp, HD, 0);
  qf[1] = load_frag(Qp, HD, 32);

  v8f o[4];
#pragma unroll
  for (int ht = 0; ht < 4; ++ht) o[ht] = (v8f){};
  float mrow[8], lrow[8];
#pragma unroll
  for (int i = 0; i < 8; ++i) { mrow[i] = -3.0e38f; lrow[i] = 0.f; }

  const int hi8 = (lane >> 4) << 3;
  const int col = lane & 15;
  _Float16 (*pw)[64] = Ps[wave];

  for (int kb = 0; kb < SEQ; kb += 64) {
    // prefetch next key/value blocks into cache (global_prefetch_b8)
    if (kb + 64 < SEQ) {
      __builtin_prefetch(Kp + (size_t)(kb + 64) * HD + lane * 16, 0, 1);
      __builtin_prefetch(Vp + (size_t)lane * SEQ + kb + 64, 0, 1);
    }

    // ---- scores: 4 tiles of 16 keys, contraction over Hd=64 (2 wmma each)
    v8f sc[4];
#pragma unroll
    for (int nt = 0; nt < 4; ++nt) {
      const _Float16* Kt = Kp + (size_t)(kb + nt * 16) * HD;
      v16h k0 = load_frag(Kt, HD, 0);
      v16h k1 = load_frag(Kt, HD, 32);
      sc[nt] = (v8f){};
      sc[nt] = wmma_f16(qf[0], k0, sc[nt]);
      sc[nt] = wmma_f16(qf[1], k1, sc[nt]);
#pragma unroll
      for (int i = 0; i < 8; ++i) sc[nt][i] *= SCALE_LOG2E;
    }

    // ---- online softmax per row (rows i / i+8 split across lane halves)
#pragma unroll
    for (int i = 0; i < 8; ++i) {
      float mx = fmaxf(fmaxf(sc[0][i], sc[1][i]), fmaxf(sc[2][i], sc[3][i]));
#pragma unroll
      for (int sft = 1; sft < 16; sft <<= 1)
        mx = fmaxf(mx, __shfl_xor(mx, sft, 32));
      const float mnew = fmaxf(mrow[i], mx);
      const float al   = exp2f(mrow[i] - mnew);
      float rs = 0.f;
#pragma unroll
      for (int nt = 0; nt < 4; ++nt) {
        const float p = exp2f(sc[nt][i] - mnew);
        sc[nt][i] = p;
        rs += p;
      }
#pragma unroll
      for (int sft = 1; sft < 16; sft <<= 1)
        rs += __shfl_xor(rs, sft, 32);
      lrow[i] = lrow[i] * al + rs;
      mrow[i] = mnew;
#pragma unroll
      for (int ht = 0; ht < 4; ++ht) o[ht][i] *= al;
    }

    // ---- bounce P through per-wave LDS to re-layout C->A fragments
#pragma unroll
    for (int nt = 0; nt < 4; ++nt)
#pragma unroll
      for (int i = 0; i < 8; ++i)
        pw[hi8 + i][nt * 16 + col] = (_Float16)sc[nt][i];

    v16h pf0 = load_frag(&pw[0][0], 64, 0);
    v16h pf1 = load_frag(&pw[0][0], 64, 32);

    // ---- O += P @ V  (B operand = V^T rows = hd, contraction over keys)
#pragma unroll
    for (int ht = 0; ht < 4; ++ht) {
      const _Float16* Vb = Vp + (size_t)(ht * 16) * SEQ + kb;
      v16h v0 = load_frag(Vb, SEQ, 0);
      v16h v1 = load_frag(Vb, SEQ, 32);
      o[ht] = wmma_f16(pf0, v0, o[ht]);
      o[ht] = wmma_f16(pf1, v1, o[ht]);
    }
  }

  // ---- normalize, store f16 as [B,S,H*Hd] (GEMM layout for out-proj)
#pragma unroll
  for (int i = 0; i < 8; ++i) {
    const float inv = 1.0f / lrow[i];
    const int qi = qBase + hi8 + i;
#pragma unroll
    for (int ht = 0; ht < 4; ++ht) {
      const int hd = ht * 16 + col;
      Ah[((size_t)(b_ * SEQ + qi)) * EMB + h * HD + hd] =
          (_Float16)(o[ht][i] * inv);
    }
  }
}

// ---------------------------------------------------------------------------
// Kernel 3: output projection.  Y = Ah(f16) @ Wo^T + bo, fp32 out.
// X tile staged via the Tensor Data Mover (async DMA, TENSORcnt) when the
// builtin is available; W tile staged manually (needs f32->f16 convert).
// ---------------------------------------------------------------------------
__global__ __launch_bounds__(256) void out_proj_kernel(
    const _Float16* __restrict__ Xh, const float* __restrict__ Wo,
    const float* __restrict__ bo, float* __restrict__ Y)
{
  __shared__ alignas(16) _Float16 Xs[BM][BK];
  __shared__ alignas(16) _Float16 Ws[BN][BK];

  const int tid  = threadIdx.x;
  const int wave = tid >> 5;
  const int lane = tid & 31;
  const int m0 = blockIdx.x * BM;
  const int n0 = blockIdx.y * BN;
  const int wm = (wave >> 2) * 64;
  const int wn = (wave & 3) * 32;

#ifdef HAVE_TDM
  // LDS byte offset of Xs: generic LDS address maps to LDS offset via addr[31:0]
  const unsigned xs_lds = (unsigned)(uintptr_t)(&Xs[0][0]);
#endif

  v8f acc[4][2];
#pragma unroll
  for (int mt = 0; mt < 4; ++mt)
#pragma unroll
    for (int nt = 0; nt < 2; ++nt) acc[mt][nt] = (v8f){};

  const int r = tid >> 1;
  const int c = (tid & 1) * 16;

  for (int kt = 0; kt < EMB; kt += BK) {
    // ---- X tile: TDM async 2D tile load (128 x 32 f16, row stride EMB)
#ifdef HAVE_TDM
    tdm_load_tile_f16(Xh + (size_t)m0 * EMB + kt, xs_lds,
                      /*tile_rows=*/BM, /*tile_cols=*/BK,
                      /*row_len=*/EMB, /*tensor_rows=*/MTOT,
                      /*row_stride=*/EMB);
#else
    {
      const v8h* sx = (const v8h*)(Xh + (size_t)(m0 + r) * EMB + kt + c);
      *(v8h*)&Xs[r][c]     = sx[0];
      *(v8h*)&Xs[r][c + 8] = sx[1];
    }
#endif
    // ---- W tile: manual load + f32->f16 convert
    {
      const float4* sw = (const float4*)(Wo + (size_t)(n0 + r) * EMB + kt + c);
#pragma unroll
      for (int i = 0; i < 2; ++i) {
        float4 f0 = sw[2 * i], f1 = sw[2 * i + 1];
        v8h hp;
        hp[0] = (_Float16)f0.x; hp[1] = (_Float16)f0.y;
        hp[2] = (_Float16)f0.z; hp[3] = (_Float16)f0.w;
        hp[4] = (_Float16)f1.x; hp[5] = (_Float16)f1.y;
        hp[6] = (_Float16)f1.z; hp[7] = (_Float16)f1.w;
        *(v8h*)&Ws[r][c + i * 8] = hp;
      }
    }
#ifdef HAVE_TDM
    __builtin_amdgcn_s_wait_tensorcnt(0);
#endif
    __syncthreads();

    v16h bfr[2];
#pragma unroll
    for (int nt = 0; nt < 2; ++nt)
      bfr[nt] = load_frag(&Ws[wn + nt * 16][0], BK, 0);
#pragma unroll
    for (int mt = 0; mt < 4; ++mt) {
      v16h a = load_frag(&Xs[wm + mt * 16][0], BK, 0);
#pragma unroll
      for (int nt = 0; nt < 2; ++nt)
        acc[mt][nt] = wmma_f16(a, bfr[nt], acc[mt][nt]);
    }
    __syncthreads();
  }

  const int hi8 = (lane >> 4) << 3;
  const int col = lane & 15;
#pragma unroll
  for (int mt = 0; mt < 4; ++mt) {
#pragma unroll
    for (int nt = 0; nt < 2; ++nt) {
      const int n = n0 + wn + nt * 16 + col;
      const float bval = bo[n];
#pragma unroll
      for (int i = 0; i < 8; ++i) {
        const int m = m0 + wm + mt * 16 + hi8 + i;
        Y[(size_t)m * EMB + n] = acc[mt][nt][i] + bval;
      }
    }
  }
}

// ---------------------------------------------------------------------------
extern "C" void kernel_launch(void* const* d_in, const int* in_sizes, int n_in,
                              void* d_out, int out_size, void* d_ws,
                              size_t ws_size, hipStream_t stream) {
  const float* q  = (const float*)d_in[0];
  const float* k  = (const float*)d_in[1];
  const float* v  = (const float*)d_in[2];
  const float* Wq = (const float*)d_in[3];
  const float* bq = (const float*)d_in[4];
  const float* Wk = (const float*)d_in[5];
  const float* bk = (const float*)d_in[6];
  const float* Wv = (const float*)d_in[7];
  const float* bv = (const float*)d_in[8];
  const float* Wo = (const float*)d_in[9];
  const float* bo = (const float*)d_in[10];
  float* out = (float*)d_out;

  const size_t mat = (size_t)MTOT * EMB;
  _Float16* Qh = (_Float16*)d_ws;
  _Float16* Kh = Qh + mat;
  _Float16* Vt = Kh + mat;
  _Float16* Ah = Vt + mat;

  qkv_proj_kernel<<<dim3(MTOT / BM, EMB / BN, 3), 256, 0, stream>>>(
      q, k, v, Wq, bq, Wk, bk, Wv, bv, Qh, Kh, Vt);

  attn_kernel<<<dim3(SEQ / 128, BATCH * HEADS), 256, 0, stream>>>(Qh, Kh, Vt,
                                                                  Ah);

  out_proj_kernel<<<dim3(MTOT / BM, EMB / BN), 256, 0, stream>>>(Ah, Wo, bo,
                                                                 out);
}
